// GptOssExperts_76656576299580
// MI455X (gfx1250) — matrix operand order repaired and checked by reading, not compiled, not run on hardware
//
#include <hip/hip_runtime.h>
#include <hip/hip_bf16.h>

// ---------------------------------------------------------------------------
// GPT-OSS MoE experts (grouped SwiGLU MLP) for gfx1250 / MI455X.
// fp32 inputs converted to bf16 in registers, compute on
// v_wmma_f32_16x16x32_bf16 (f32 accumulate).  Two kernels:
//   1) gate/up fused GEMM + SwiGLU  -> bf16 activated[T,I] in workspace
//   2) down GEMM + bias             -> fp32 out[T,H]
// Tile: BM=256 x BN=64 x BK=32, 8 waves as 4(M) x 2(N), wave tile 64x32
// => 1 ds_load_b128 per WMMA (A and B frag cost each amortized).
// ---------------------------------------------------------------------------

#define E_ 8
#define H_ 2048
#define I_ 4096
#define T_ 8192

#define BM 256          // tokens per block tile (1024/expert divides evenly)
#define BN 64           // output columns per block tile
#define BK 32           // K depth per LDS stage (one WMMA K)
#define LDA 40          // padded LDS stride (bf16 elems) -> 80B, 16B aligned
#define LDB 40

typedef __bf16 bf16_t;
typedef bf16_t v16bf __attribute__((ext_vector_type(16)));
typedef float  v8f   __attribute__((ext_vector_type(8)));

__device__ __forceinline__ bf16_t f2bf(float f) { return (bf16_t)f; }

// pack 4 floats -> 4 bf16 and store 8 bytes to LDS
__device__ __forceinline__ void st4bf(bf16_t* p, float a, float b, float c, float d) {
    bf16_t v[4] = {f2bf(a), f2bf(b), f2bf(c), f2bf(d)};
    *reinterpret_cast<uint2*>(p) = *reinterpret_cast<const uint2*>(v);
}

// --- WMMA fragment gathers (match CDNA5 ISA 16-bit A / B VGPR layouts) -----
// A tile in LDS: [16 rows][K=32] row-major (stride LDA).
// lane<16: m=lane, K in {0..7, 16..23}; lane>=16: m=lane-16, K in {8..15, 24..31}
__device__ __forceinline__ v16bf ld_frag_a(const bf16_t* base, int lane) {
    const bf16_t* p = base + (lane & 15) * LDA + ((lane >> 4) << 3);
    v16bf a;
    *reinterpret_cast<uint4*>(&a)       = *reinterpret_cast<const uint4*>(p);
    *(reinterpret_cast<uint4*>(&a) + 1) = *reinterpret_cast<const uint4*>(p + 16);
    return a;
}
// B tile in LDS (transposed): [16 cols n][K=32] row-major (stride LDB).
// lanes 0-15: n=lane, K in {0..7, 16..23}; lanes 16-31: K in {8..15, 24..31}
__device__ __forceinline__ v16bf ld_frag_b(const bf16_t* base, int lane) {
    const bf16_t* p = base + (lane & 15) * LDB + ((lane >> 4) << 3);
    v16bf b;
    *reinterpret_cast<uint4*>(&b)       = *reinterpret_cast<const uint4*>(p);
    *(reinterpret_cast<uint4*>(&b) + 1) = *reinterpret_cast<const uint4*>(p + 16);
    return b;
}

// --- global -> register prefetch helpers -----------------------------------
// x tile: BM x BK fp32, coalesced float4 loads (8 per thread)
__device__ __forceinline__ void ldg_x(float4 rA[8], const float* __restrict__ x,
                                      int mBase, int k0, int tid) {
#pragma unroll
    for (int i = 0; i < 8; ++i) {
        int f = tid + i * 256;          // float4 slot in tile (8 per row)
        int row = f >> 3, c4 = f & 7;
        rA[i] = *reinterpret_cast<const float4*>(
            x + (size_t)(mBase + row) * H_ + k0 + (c4 << 2));
    }
}
__device__ __forceinline__ void sts_x(const float4 rA[8], bf16_t* ldsA, int tid) {
#pragma unroll
    for (int i = 0; i < 8; ++i) {
        int f = tid + i * 256;
        int row = f >> 3, c4 = f & 7;
        st4bf(ldsA + row * LDA + (c4 << 2), rA[i].x, rA[i].y, rA[i].z, rA[i].w);
    }
}
// weight tile: BK x BN fp32 (row stride ldw), gathered K-major into registers
__device__ __forceinline__ void ldg_w(float rB[8], const float* __restrict__ w,
                                      int k0, int nBase, int ldw, int tid) {
    int n  = tid & 63;
    int kq = (tid >> 6) << 2;           // 0,4,8,12
#pragma unroll
    for (int j = 0; j < 4; ++j) {
        rB[j]     = w[(size_t)(k0 + kq + j) * ldw + nBase + n];
        rB[4 + j] = w[(size_t)(k0 + 16 + kq + j) * ldw + nBase + n];
    }
}
__device__ __forceinline__ void sts_w(const float rB[8], bf16_t* ldsB, int tid) {
    int n  = tid & 63;
    int kq = (tid >> 6) << 2;
    st4bf(ldsB + n * LDB + kq,      rB[0], rB[1], rB[2], rB[3]);
    st4bf(ldsB + n * LDB + 16 + kq, rB[4], rB[5], rB[6], rB[7]);
}

__device__ __forceinline__ int expert_of(const int* __restrict__ offs, int tok) {
    int e = 0;
    while (e < E_ - 1 && offs[e] <= tok) ++e;   // searchsorted(side="right")
    return e;
}

// ---------------------------------------------------------------------------
// Kernel 1: gate & up GEMMs fused + SwiGLU -> bf16 activated[T, I]
// ---------------------------------------------------------------------------
__global__ __launch_bounds__(256)
void moe_gate_up_kernel(const float* __restrict__ x,
                        const int* __restrict__ offs,
                        const float* __restrict__ gate_w,
                        const float* __restrict__ up_w,
                        const float* __restrict__ gate_b,
                        const float* __restrict__ up_b,
                        bf16_t* __restrict__ act) {
    __shared__ bf16_t ldsA[BM * LDA];
    __shared__ bf16_t ldsG[BN * LDB];
    __shared__ bf16_t ldsU[BN * LDB];

    const int tid  = threadIdx.x;
    const int lane = tid & 31;
    const int wid  = tid >> 5;
    const int wm   = wid & 3;                 // wave M slot (x64 rows)
    const int wn   = wid >> 2;                // wave N slot (x32 cols)
    const int mBase = blockIdx.y * BM;
    const int nBase = blockIdx.x * BN;

    const int e = expert_of(offs, mBase);
    const float* gw = gate_w + (size_t)e * H_ * I_;
    const float* uw = up_w   + (size_t)e * H_ * I_;

    v8f accG[4][2] = {{{0}}};
    v8f accU[4][2] = {{{0}}};

    float4 rA[8]; float rG[8]; float rU[8];
    ldg_x(rA, x, mBase, 0, tid);
    ldg_w(rG, gw, 0, nBase, I_, tid);
    ldg_w(rU, uw, 0, nBase, I_, tid);

    for (int k0 = 0; k0 < H_; k0 += BK) {
        sts_x(rA, ldsA, tid);
        sts_w(rG, ldsG, tid);
        sts_w(rU, ldsU, tid);
        __syncthreads();
        if (k0 + BK < H_) {                    // prefetch next stage
            ldg_x(rA, x, mBase, k0 + BK, tid);
            ldg_w(rG, gw, k0 + BK, nBase, I_, tid);
            ldg_w(rU, uw, k0 + BK, nBase, I_, tid);
        }
        v16bf af[4], gf[2], uf[2];
#pragma unroll
        for (int mi = 0; mi < 4; ++mi)
            af[mi] = ld_frag_a(ldsA + (wm * 64 + mi * 16) * LDA, lane);
#pragma unroll
        for (int ni = 0; ni < 2; ++ni) {
            gf[ni] = ld_frag_b(ldsG + (wn * 32 + ni * 16) * LDB, lane);
            uf[ni] = ld_frag_b(ldsU + (wn * 32 + ni * 16) * LDB, lane);
        }
#pragma unroll
        for (int mi = 0; mi < 4; ++mi)
#pragma unroll
            for (int ni = 0; ni < 2; ++ni) {
                accG[mi][ni] = __builtin_amdgcn_wmma_f32_16x16x32_bf16(
                    false, af[mi], false, gf[ni], (short)0, accG[mi][ni], false, false);
                accU[mi][ni] = __builtin_amdgcn_wmma_f32_16x16x32_bf16(
                    false, af[mi], false, uf[ni], (short)0, accU[mi][ni], false, false);
            }
        __syncthreads();
    }

    // Epilogue: bias + clamped SwiGLU, store bf16 activations.
    const int nCol  = lane & 15;
    const int rowHi = (lane >> 4) << 3;       // C/D layout: M = r + 8*(lane>=16)
#pragma unroll
    for (int ni = 0; ni < 2; ++ni) {
        int n = nBase + wn * 32 + ni * 16 + nCol;
        float gb = gate_b[e * I_ + n];
        float ub = up_b[e * I_ + n];
#pragma unroll
        for (int mi = 0; mi < 4; ++mi) {
#pragma unroll
            for (int r = 0; r < 8; ++r) {
                int m = mBase + wm * 64 + mi * 16 + rowHi + r;
                float g = accG[mi][ni][r] + gb;
                float u = accU[mi][ni][r] + ub;
                g = fminf(g, 7.0f);
                u = fminf(fmaxf(u, -7.0f), 7.0f);
                float glu = g / (1.0f + __expf(-1.702f * g));
                act[(size_t)m * I_ + n] = f2bf((u + 1.0f) * glu);
            }
        }
    }
}

// ---------------------------------------------------------------------------
// Kernel 2: down GEMM: out[T,H] = act[T,I](bf16) @ down_w[e] + down_b
// ---------------------------------------------------------------------------
__global__ __launch_bounds__(256)
void moe_down_kernel(const bf16_t* __restrict__ act,
                     const int* __restrict__ offs,
                     const float* __restrict__ down_w,
                     const float* __restrict__ down_b,
                     float* __restrict__ out) {
    __shared__ bf16_t ldsA[BM * LDA];
    __shared__ bf16_t ldsB[BN * LDB];

    const int tid  = threadIdx.x;
    const int lane = tid & 31;
    const int wid  = tid >> 5;
    const int wm   = wid & 3;
    const int wn   = wid >> 2;
    const int mBase = blockIdx.y * BM;
    const int nBase = blockIdx.x * BN;

    const int e = expert_of(offs, mBase);
    const float* dw = down_w + (size_t)e * I_ * H_;

    v8f acc[4][2] = {{{0}}};

    uint4 rA[4]; float rB[8];
    // bf16 A tile: BM x BK, pure b128 copies (4 per thread)
#pragma unroll
    for (int i = 0; i < 4; ++i) {
        int f = tid + i * 256, row = f >> 2, c = f & 3;
        rA[i] = *reinterpret_cast<const uint4*>(
            act + (size_t)(mBase + row) * I_ + 0 + (c << 3));
    }
    ldg_w(rB, dw, 0, nBase, H_, tid);

    for (int k0 = 0; k0 < I_; k0 += BK) {
#pragma unroll
        for (int i = 0; i < 4; ++i) {
            int f = tid + i * 256, row = f >> 2, c = f & 3;
            *reinterpret_cast<uint4*>(ldsA + row * LDA + (c << 3)) = rA[i];
        }
        sts_w(rB, ldsB, tid);
        __syncthreads();
        if (k0 + BK < I_) {
#pragma unroll
            for (int i = 0; i < 4; ++i) {
                int f = tid + i * 256, row = f >> 2, c = f & 3;
                rA[i] = *reinterpret_cast<const uint4*>(
                    act + (size_t)(mBase + row) * I_ + (k0 + BK) + (c << 3));
            }
            ldg_w(rB, dw, k0 + BK, nBase, H_, tid);
        }
        v16bf af[4], bf[2];
#pragma unroll
        for (int mi = 0; mi < 4; ++mi)
            af[mi] = ld_frag_a(ldsA + (wm * 64 + mi * 16) * LDA, lane);
#pragma unroll
        for (int ni = 0; ni < 2; ++ni)
            bf[ni] = ld_frag_b(ldsB + (wn * 32 + ni * 16) * LDB, lane);
#pragma unroll
        for (int mi = 0; mi < 4; ++mi)
#pragma unroll
            for (int ni = 0; ni < 2; ++ni)
                acc[mi][ni] = __builtin_amdgcn_wmma_f32_16x16x32_bf16(
                    false, af[mi], false, bf[ni], (short)0, acc[mi][ni], false, false);
        __syncthreads();
    }

    const int nCol  = lane & 15;
    const int rowHi = (lane >> 4) << 3;
#pragma unroll
    for (int ni = 0; ni < 2; ++ni) {
        int n = nBase + wn * 32 + ni * 16 + nCol;
        float db = down_b[e * H_ + n];
#pragma unroll
        for (int mi = 0; mi < 4; ++mi) {
#pragma unroll
            for (int r = 0; r < 8; ++r) {
                int m = mBase + wm * 64 + mi * 16 + rowHi + r;
                out[(size_t)m * H_ + n] = acc[mi][ni][r] + db;
            }
        }
    }
}

// ---------------------------------------------------------------------------
extern "C" void kernel_launch(void* const* d_in, const int* in_sizes, int n_in,
                              void* d_out, int out_size, void* d_ws, size_t ws_size,
                              hipStream_t stream) {
    const float* x      = (const float*)d_in[0];
    const int*   offs   = (const int*)d_in[1];
    const float* gate_w = (const float*)d_in[2];
    const float* up_w   = (const float*)d_in[3];
    const float* down_w = (const float*)d_in[4];
    const float* gate_b = (const float*)d_in[5];
    const float* up_b   = (const float*)d_in[6];
    const float* down_b = (const float*)d_in[7];

    bf16_t* act = (bf16_t*)d_ws;            // T * I bf16 = 64 MB

    dim3 blk(256);
    dim3 g1(I_ / BN, T_ / BM);              // 64 x 32 blocks
    moe_gate_up_kernel<<<g1, blk, 0, stream>>>(x, offs, gate_w, up_w,
                                               gate_b, up_b, act);
    dim3 g2(H_ / BN, T_ / BM);              // 32 x 32 blocks
    moe_down_kernel<<<g2, blk, 0, stream>>>(act, offs, down_w, down_b,
                                            (float*)d_out);
}
